// VoxelSetAbstraction_63127429317326
// MI455X (gfx1250) — matrix operand order repaired and checked by reading, not compiled, not run on hardware
//
#include <hip/hip_runtime.h>

// ---------------------------------------------------------------------------
// VoxelSetAbstraction for MI455X (gfx1250, wave32).
//   1) fps_kernel        : persistent-block farthest point sampling (B blocks)
//   2) bev_sample_kernel : bilinear BEV gather -> feat[:, 0:256]
//   3) group_mlp_kernel  : wave-per-keypoint ballot radius scan + tiny MLPs
//                          + lane max-pool -> feat[:, 256:288]
//   4) fuse_gemm_kernel  : (4096x288)@(288x128) via v_wmma_f32_16x16x32_f16,
//                          LDS-staged f16 weight tile, fully unrolled K chain,
//                          fused scale/bias/relu epilogue -> d_out
// ---------------------------------------------------------------------------

#define BB     2
#define NPTS   16384
#define KPTS   2048
#define CBEV   256
#define HH     128
#define WW     128
#define CIN    288
#define COUT   128

typedef __attribute__((ext_vector_type(16))) _Float16 v16h;
typedef __attribute__((ext_vector_type(8)))  float    v8f;

// ---------------- Kernel 1: farthest point sampling ------------------------
#define FPS_T 1024
#define PPT   (NPTS / FPS_T)   // 16 points per thread, register-resident

__global__ __launch_bounds__(FPS_T)
void fps_kernel(const float* __restrict__ points, float* __restrict__ kp_xyz) {
  const int b = blockIdx.x;
  const int t = threadIdx.x;
  const int lane = t & 31;
  const int wv = t >> 5;
  const float* base = points + (size_t)b * NPTS * 4;

  float px[PPT], py[PPT], pz[PPT], dist[PPT];
#pragma unroll
  for (int j = 0; j < PPT; ++j) {
    int p = j * FPS_T + t;
    px[j] = base[p * 4 + 0];
    py[j] = base[p * 4 + 1];
    pz[j] = base[p * 4 + 2];
    dist[j] = 1e10f;
  }

  __shared__ float sfx, sfy, sfz;
  __shared__ int   sfar;
  __shared__ float swd[32];
  __shared__ int   swi[32];

  int far = 0;  // reference: far0 = argmax(zeros) = 0
  for (int it = 0; it < KPTS; ++it) {
    if (t == 0) {
      float fx = base[far * 4 + 0];
      float fy = base[far * 4 + 1];
      float fz = base[far * 4 + 2];
      sfx = fx; sfy = fy; sfz = fz;
      float* kp = kp_xyz + ((size_t)b * KPTS + it) * 3;
      kp[0] = fx; kp[1] = fy; kp[2] = fz;   // keypoint emitted is pre-update far
    }
    __syncthreads();
    float fx = sfx, fy = sfy, fz = sfz;
    float bd = -1.0f; int bi = 0x7fffffff;
#pragma unroll
    for (int j = 0; j < PPT; ++j) {
      float dx = px[j] - fx, dy = py[j] - fy, dz = pz[j] - fz;
      float d = dx * dx + dy * dy + dz * dz;
      dist[j] = fminf(dist[j], d);
      if (dist[j] > bd) { bd = dist[j]; bi = j * FPS_T + t; }
    }
    // wave32 argmax butterfly (first-max-index tiebreak like jnp.argmax)
#pragma unroll
    for (int off = 16; off > 0; off >>= 1) {
      float od = __shfl_xor(bd, off);
      int   oi = __shfl_xor(bi, off);
      if (od > bd || (od == bd && oi < bi)) { bd = od; bi = oi; }
    }
    if (lane == 0) { swd[wv] = bd; swi[wv] = bi; }
    __syncthreads();
    if (t < 32) {
      bd = swd[t]; bi = swi[t];
#pragma unroll
      for (int off = 16; off > 0; off >>= 1) {
        float od = __shfl_xor(bd, off);
        int   oi = __shfl_xor(bi, off);
        if (od > bd || (od == bd && oi < bi)) { bd = od; bi = oi; }
      }
      if (t == 0) sfar = bi;
    }
    __syncthreads();
    far = sfar;
  }
}

// ---------------- Kernel 2: bilinear BEV sampling --------------------------
__global__ __launch_bounds__(256)
void bev_sample_kernel(const float* __restrict__ bev,
                       const float* __restrict__ kp_xyz,
                       float* __restrict__ feat) {
  int gid = blockIdx.x * 256 + threadIdx.x;     // BB*KPTS*CBEV threads
  int c  = gid & (CBEV - 1);
  int kf = gid >> 8;                            // b*KPTS + k
  const float* kp = kp_xyz + (size_t)kf * 3;
  float x = (kp[0] + 51.2f) * 1.25f;            // (x - pc_min)/voxel/stride
  float y = (kp[1] + 51.2f) * 1.25f;
  int xf = (int)floorf(x), yf = (int)floorf(y);
  int x0 = min(max(xf,     0), WW - 1);
  int x1 = min(max(xf + 1, 0), WW - 1);
  int y0 = min(max(yf,     0), HH - 1);
  int y1 = min(max(yf + 1, 0), HH - 1);
  const float* im = bev + ((size_t)(kf >> 11) * CBEV + c) * (size_t)(HH * WW);
  float Ia = im[y0 * WW + x0], Ib = im[y1 * WW + x0];
  float Ic = im[y0 * WW + x1], Id = im[y1 * WW + x1];
  float x0f = (float)x0, x1f = (float)x1, y0f = (float)y0, y1f = (float)y1;
  float wa = (x1f - x) * (y1f - y);
  float wb = (x1f - x) * (y - y0f);
  float wc = (x - x0f) * (y1f - y);
  float wd = (x - x0f) * (y - y0f);
  feat[(size_t)kf * CIN + c] = Ia * wa + Ib * wb + Ic * wc + Id * wd;
}

// ---------------- Kernel 3: radius grouping + tiny MLPs + maxpool ----------
__global__ __launch_bounds__(256)
void group_mlp_kernel(const float* __restrict__ points,
                      const float* __restrict__ kp_xyz,
                      const float* __restrict__ w1, const float* __restrict__ s1,
                      const float* __restrict__ b1, const float* __restrict__ w2,
                      const float* __restrict__ s2, const float* __restrict__ b2,
                      float* __restrict__ feat) {
  __shared__ float wg[768];        // [w1:128][s1:32][b1:32][w2:512][s2:32][b2:32]
  __shared__ int   sidx1[8][16];
  __shared__ int   sidx2[8][32];
  int t = threadIdx.x;
  for (int i = t; i < 128; i += 256) wg[i] = w1[i];
  for (int i = t; i < 512; i += 256) wg[192 + i] = w2[i];
  if (t < 32) {
    wg[128 + t] = s1[t]; wg[160 + t] = b1[t];
    wg[704 + t] = s2[t]; wg[736 + t] = b2[t];
  }
  __syncthreads();

  int wv = t >> 5, lane = t & 31;
  int kf = blockIdx.x * 8 + wv;                 // keypoint flat id, 0..4095
  int b  = kf >> 11;
  const float* base = points + (size_t)b * NPTS * 4;
  const float* kp = kp_xyz + (size_t)kf * 3;
  float kx = kp[0], ky = kp[1], kz = kp[2];

  // Ordered scan: first 16 indices inside r=0.4, first 32 inside r=0.8.
  int c1 = 0, c2 = 0;
  for (int s = 0; s < NPTS && (c1 < 16 || c2 < 32); s += 32) {
    int p = s + lane;
    float dx = base[p * 4 + 0] - kx;
    float dy = base[p * 4 + 1] - ky;
    float dz = base[p * 4 + 2] - kz;
    float d = dx * dx + dy * dy + dz * dz;
    bool in1 = d < 0.16f, in2 = d < 0.64f;
    unsigned m1 = (unsigned)__ballot(in1);
    unsigned m2 = (unsigned)__ballot(in2);
    unsigned lt = (1u << lane) - 1u;
    if (in2) { int pos = c2 + __popc(m2 & lt); if (pos < 32) sidx2[wv][pos] = p; }
    if (in1) { int pos = c1 + __popc(m1 & lt); if (pos < 16) sidx1[wv][pos] = p; }
    c2 = min(32, c2 + __popc(m2));
    c1 = min(16, c1 + __popc(m1));
  }

#pragma unroll
  for (int br = 0; br < 2; ++br) {
    const int ns  = (br == 0) ? 16 : 32;
    const int cnt = (br == 0) ? c1 : c2;
    bool active = lane < ns;
    int id;
    if (br == 0) id = (lane < cnt) ? sidx1[wv][lane] : ((cnt > 0) ? sidx1[wv][0] : 0);
    else         id = (lane < cnt) ? sidx2[wv][lane] : ((cnt > 0) ? sidx2[wv][0] : 0);
    float g0 = base[id * 4 + 0] - kx;
    float g1 = base[id * 4 + 1] - ky;
    float g2 = base[id * 4 + 2] - kz;
    float g3 = base[id * 4 + 3];
    if (cnt == 0) { g0 = 0.f; g1 = 0.f; g2 = 0.f; g3 = 0.f; }  // empty group => g=0

    const float* W1 = &wg[br * 64];
    const float* S1 = &wg[128 + br * 16];
    const float* B1 = &wg[160 + br * 16];
    const float* W2 = &wg[192 + br * 256];
    const float* S2 = &wg[704 + br * 16];
    const float* B2 = &wg[736 + br * 16];

    float h1[16];
#pragma unroll
    for (int d = 0; d < 16; ++d) {
      float a = g0 * W1[d] + g1 * W1[16 + d] + g2 * W1[32 + d] + g3 * W1[48 + d];
      h1[d] = fmaxf(a * S1[d] + B1[d], 0.0f);
    }
#pragma unroll
    for (int d = 0; d < 16; ++d) {
      float a = 0.0f;
#pragma unroll
      for (int c = 0; c < 16; ++c) a += h1[c] * W2[c * 16 + d];
      a = fmaxf(a * S2[d] + B2[d], 0.0f);
      float v = active ? a : -1e30f;          // exclude lanes beyond nsample
#pragma unroll
      for (int off = 16; off > 0; off >>= 1) v = fmaxf(v, __shfl_xor(v, off));
      if (lane == 0) feat[(size_t)kf * CIN + CBEV + br * 16 + d] = v;
    }
  }
}

// ---------------- Kernel 4: fused GEMM via WMMA ----------------------------
// (4096 x 288) @ (288 x 128) -> relu(.*s + b).
// Block = one N tile (16 cols) shared by 8 waves covering 8 M tiles.
// Weight tile converted to f16 once into LDS; fragments read as v16h
// (32B-aligned -> ds_load_b128 pairs). K chain fully unrolled: 9 WMMAs.
__global__ __launch_bounds__(256)
void fuse_gemm_kernel(const float* __restrict__ feat, const float* __restrict__ fw,
                      const float* __restrict__ fs, const float* __restrict__ fb,
                      float* __restrict__ out) {
  __shared__ alignas(32) _Float16 blds[16 * CIN];   // [n within tile][k]
  int t = threadIdx.x;
  int ntile = blockIdx.x & 7;
  int mbase = (blockIdx.x >> 3) * 8;

  // Stage B tile: fw[k][ntile*16 + n] -> blds[n*CIN + k], f32 -> f16.
  for (int i = t; i < 16 * CIN; i += 256) {
    int k = i >> 4, n = i & 15;                 // coalesced over n
    blds[n * CIN + k] = (_Float16)fw[(size_t)k * COUT + ntile * 16 + n];
  }
  __syncthreads();

  int wv = t >> 5, lane = t & 31;
  int mtile = mbase + wv;
  int hi = lane >> 4, lo = lane & 15;
  int m = mtile * 16 + lo;                      // A row owned by this lane
  int n = ntile * 16 + lo;                      // C/D column owned by this lane
  const float4* arow4 = reinterpret_cast<const float4*>(feat + (size_t)m * CIN);
  __builtin_prefetch(feat + (size_t)m * CIN, 0, 0);   // global_prefetch_b8

  v8f acc = {};
#pragma unroll
  for (int kk = 0; kk < CIN; kk += 32) {        // 288/32 = 9 WMMA steps
    // A fragment, 16-bit 16x32 layout (contiguous K, matching B):
    //   lanes 0-15: K 0..15 ; lanes 16-31: K 16..31
    float4 a0 = arow4[(kk + 16 * hi) >> 2];
    float4 a1 = arow4[((kk + 16 * hi) >> 2) + 1];
    float4 a2 = arow4[((kk + 16 * hi) >> 2) + 2];
    float4 a3 = arow4[((kk + 16 * hi) >> 2) + 3];
    v16h a;
    a[0]  = (_Float16)a0.x; a[1]  = (_Float16)a0.y;
    a[2]  = (_Float16)a0.z; a[3]  = (_Float16)a0.w;
    a[4]  = (_Float16)a1.x; a[5]  = (_Float16)a1.y;
    a[6]  = (_Float16)a1.z; a[7]  = (_Float16)a1.w;
    a[8]  = (_Float16)a2.x; a[9]  = (_Float16)a2.y;
    a[10] = (_Float16)a2.z; a[11] = (_Float16)a2.w;
    a[12] = (_Float16)a3.x; a[13] = (_Float16)a3.y;
    a[14] = (_Float16)a3.z; a[15] = (_Float16)a3.w;
    // B fragment, 32x16: rows K striped 16 per half-wave, N = lane&15.
    // blds offset bytes = lo*576 + kk*2 + hi*32 -> all multiples of 32: aligned.
    v16h bm = *reinterpret_cast<const v16h*>(&blds[lo * CIN + kk + 16 * hi]);
    acc = __builtin_amdgcn_wmma_f32_16x16x32_f16(
        /*neg_a=*/false, a, /*neg_b=*/false, bm,
        /*c_mod=*/(short)0, acc, /*reuse_a=*/false, /*reuse_b=*/false);
  }

  float sc = fs[n], bi = fb[n];
#pragma unroll
  for (int r = 0; r < 8; ++r) {                 // C/D: VGPR r -> M = 8*hi + r
    int row = mtile * 16 + 8 * hi + r;
    out[(size_t)row * COUT + n] = fmaxf(acc[r] * sc + bi, 0.0f);
  }
}

// ---------------- launcher -------------------------------------------------
extern "C" void kernel_launch(void* const* d_in, const int* in_sizes, int n_in,
                              void* d_out, int out_size, void* d_ws, size_t ws_size,
                              hipStream_t stream) {
  const float* points = (const float*)d_in[0];
  const float* bev    = (const float*)d_in[1];
  const float* sa_w1  = (const float*)d_in[2];
  const float* sa_s1  = (const float*)d_in[3];
  const float* sa_b1  = (const float*)d_in[4];
  const float* sa_w2  = (const float*)d_in[5];
  const float* sa_s2  = (const float*)d_in[6];
  const float* sa_b2  = (const float*)d_in[7];
  const float* fuse_w = (const float*)d_in[8];
  const float* fuse_s = (const float*)d_in[9];
  const float* fuse_b = (const float*)d_in[10];
  float* out = (float*)d_out;

  float* feat   = (float*)d_ws;                          // [4096][288] f32
  float* kp_xyz = feat + (size_t)BB * KPTS * CIN;        // [2][2048][3] f32

  fps_kernel<<<BB, FPS_T, 0, stream>>>(points, kp_xyz);
  bev_sample_kernel<<<(BB * KPTS * CBEV) / 256, 256, 0, stream>>>(bev, kp_xyz, feat);
  group_mlp_kernel<<<(BB * KPTS) / 8, 256, 0, stream>>>(
      points, kp_xyz, sa_w1, sa_s1, sa_b1, sa_w2, sa_s2, sa_b2, feat);
  // 256 M-tiles x 8 N-tiles, 8 waves (one M-tile each) per block -> 256 blocks
  fuse_gemm_kernel<<<256, 256, 0, stream>>>(feat, fuse_w, fuse_s, fuse_b, out);
}
